// GlobalAttention_15436112462100
// MI455X (gfx1250) — matrix operand order repaired and checked by reading, not compiled, not run on hardware
//
#include <hip/hip_runtime.h>

// ---------------- problem constants (match reference) ----------------
#define B_    2
#define T_    1024
#define S_    1024
#define NH    16        // N_HEADS
#define NG    4         // N_GROUPS
#define HQ_   32
#define HV_   32
#define CINV  256
#define CVEC  128
#define FVEC  2
#define DHEAD 96        // HQ * (1 + F_VEC) combined head width
#define KX    512       // packed activation width: 256 inv + 2*128 vec
#define SOFT_SCALE 0.10206207261596575f   // 1/sqrt(96)

// ---------------- WMMA vector types (gfx1250, wave32) ----------------
typedef __attribute__((ext_vector_type(16))) __bf16 v16bf;
typedef __attribute__((ext_vector_type(8)))  __bf16 v8bf;
typedef __attribute__((ext_vector_type(8)))  float  v8f;
typedef __attribute__((ext_vector_type(4)))  int    v4i;

union V16U { v16bf v; v8bf h[2]; };

__device__ __forceinline__ unsigned short f2bf_bits(float f) {
  unsigned u = __float_as_uint(f);
  u += 0x7FFFu + ((u >> 16) & 1u);          // round-to-nearest-even
  return (unsigned short)(u >> 16);
}

// ---------------- gfx1250 async global->LDS path (probe-guarded) ------
#if defined(__has_builtin)
#  if __has_builtin(__builtin_amdgcn_global_load_async_to_lds_b128)
#    define USE_ASYNC 1
#  endif
#endif
#ifndef USE_ASYNC
#  define USE_ASYNC 0
#endif

#if USE_ASYNC
typedef __attribute__((address_space(1))) v4i* gbl_v4i_ptr;
typedef __attribute__((address_space(3))) v4i* lds_v4i_ptr;
#endif

__device__ __forceinline__ void async_copy16(const unsigned short* gp,
                                             unsigned short* lp) {
#if USE_ASYNC
  __builtin_amdgcn_global_load_async_to_lds_b128(
      (gbl_v4i_ptr)(unsigned short*)gp, (lds_v4i_ptr)lp, 0, 0);
#else
  *(v8bf*)lp = *(const v8bf*)gp;
#endif
}

__device__ __forceinline__ void wait_async_all() {
#if USE_ASYNC
#  if __has_builtin(__builtin_amdgcn_s_wait_asynccnt)
  __builtin_amdgcn_s_wait_asynccnt(0);
#  else
  asm volatile("s_wait_asynccnt 0" ::: "memory");
#  endif
#endif
}

// =====================================================================
// Prep 1: pack activations to bf16: X[bt][512] = [inv(256)|vec0(128)|vec1(128)]
// Builds both target (Xt) and source (Xs) matrices.
// =====================================================================
__global__ void prep_x_kernel(const float* __restrict__ tinv,
                              const float* __restrict__ tvec,
                              const float* __restrict__ sinv,
                              const float* __restrict__ svec,
                              unsigned short* __restrict__ Xt,
                              unsigned short* __restrict__ Xs) {
  int idx = blockIdx.x * blockDim.x + threadIdx.x;
  const int per = B_ * T_ * KX;
  if (idx >= 2 * per) return;
  int which = idx / per;                 // 0 = target, 1 = source
  int rem = idx - which * per;
  int bt = rem / KX, c = rem % KX;
  const float* inv = which ? sinv : tinv;
  const float* vec = which ? svec : tvec;
  float v = (c < CINV)
                ? inv[(size_t)bt * CINV + c]
                : vec[(size_t)(bt * FVEC + ((c - CINV) >> 7)) * CVEC +
                      ((c - CINV) & 127)];
  (which ? Xs : Xt)[(size_t)bt * KX + c] = f2bf_bits(v);
}

// =====================================================================
// Prep 2: combined, TRANSPOSED, zero-padded bf16 Q weight WqCt[n][512],
// n = H*96 + d. Softmax scale folded in. Zero pad makes one uniform
// K=512 GEMM (extra FLOPs are free on WMMA vs the load-bound alternative).
// =====================================================================
__global__ void prep_wq_kernel(const float* __restrict__ Wq_inv,
                               const float* __restrict__ Wq_vec,
                               unsigned short* __restrict__ WqCt) {
  int idx = blockIdx.x * blockDim.x + threadIdx.x;
  const int total = NH * DHEAD * KX;
  if (idx >= total) return;
  int n = idx / KX, c = idx % KX;
  int H = n / DHEAD, d = n % DHEAD;
  float v = 0.f;
  if (d < 32) {
    if (c < CINV) v = Wq_inv[c * (NH * HQ_) + H * 32 + d];
  } else {
    int f = (d - 32) >> 5, hq = (d - 32) & 31, cb = CINV + f * CVEC;
    if (c >= cb && c < cb + CVEC)
      v = Wq_vec[(c - cb) * (NH * HQ_) + H * 32 + hq];
  }
  WqCt[idx] = f2bf_bits(v * SOFT_SCALE);
}

// =====================================================================
// Prep 3: combined transposed bf16 K and V weights, n = g*96 + d.
// =====================================================================
__global__ void prep_wkv_kernel(const float* __restrict__ Wk_inv,
                                const float* __restrict__ Wk_vec,
                                const float* __restrict__ Wv_inv,
                                const float* __restrict__ Wv_vec,
                                unsigned short* __restrict__ WkCt,
                                unsigned short* __restrict__ WvCt) {
  int idx = blockIdx.x * blockDim.x + threadIdx.x;
  const int total = NG * DHEAD * KX;
  if (idx >= total) return;
  int n = idx / KX, c = idx % KX;
  int g = n / DHEAD, d = n % DHEAD;
  float vk = 0.f, vv = 0.f;
  if (d < 32) {
    if (c < CINV) {
      vk = Wk_inv[c * (NG * HQ_) + g * 32 + d];
      vv = Wv_inv[c * (NG * HV_) + g * 32 + d];
    }
  } else {
    int f = (d - 32) >> 5, hq = (d - 32) & 31, cb = CINV + f * CVEC;
    if (c >= cb && c < cb + CVEC) {
      vk = Wk_vec[(c - cb) * (NG * HQ_) + g * 32 + hq];
      vv = Wv_vec[(c - cb) * (NG * HV_) + g * 32 + hq];
    }
  }
  WkCt[idx] = f2bf_bits(vk);
  WvCt[idx] = f2bf_bits(vv);
}

// =====================================================================
// Prep 4: transposed bf16 output weights: WoInvT[256][512], WoVecT[128][512].
// =====================================================================
__global__ void prep_wo_kernel(const float* __restrict__ Wo_inv,
                               const float* __restrict__ Wo_vec,
                               unsigned short* __restrict__ WoInvT,
                               unsigned short* __restrict__ WoVecT) {
  int idx = blockIdx.x * blockDim.x + threadIdx.x;
  const int n_i = CINV * KX;
  const int n_v = CVEC * KX;
  if (idx < n_i) {
    int n = idx / KX, j = idx % KX;
    WoInvT[idx] = f2bf_bits(Wo_inv[j * CINV + n]);
  } else if (idx < n_i + n_v) {
    int k = idx - n_i;
    int n = k / KX, j = k % KX;
    WoVecT[k] = f2bf_bits(Wo_vec[j * CVEC + n]);
  }
}

// =====================================================================
// Generic bf16 WMMA GEMM: C(M x N) = A(M x 512) * B(512 x N).
// A row-major bf16 (lda); B given TRANSPOSED Bt[n][512] so B-fragments
// are contiguous 32B per-lane loads. One 16x32 output tile per wave.
// mode 0: bf16 store C[row*rowStride+col]
// mode 1: bf16 store V-transposed: Vt[((row>>10)*384 + col)*1024 + (row&1023)]
// mode 2: f32  store C[row*rowStride+col]
// =====================================================================
__global__ void __launch_bounds__(256)
gemm_bf16_kernel(const unsigned short* __restrict__ A, int lda,
                 const unsigned short* __restrict__ Bt, int tilesN,
                 void* __restrict__ Cout, int mode, int rowStride) {
  const int lane = threadIdx.x & 31;
  const int w = threadIdx.x >> 5;
  const int hi = lane >> 4, ln = lane & 15;
  const int gid = blockIdx.x * 8 + w;
  const int tM = gid / tilesN, tN = gid % tilesN;

  v8f acc[2];
#pragma unroll
  for (int nt = 0; nt < 2; ++nt)
#pragma unroll
    for (int r = 0; r < 8; ++r) acc[nt][r] = 0.f;

  const unsigned short* arow = A + (size_t)(tM * 16 + ln) * lda;
  const unsigned short* brow0 = Bt + (size_t)(tN * 32 + ln) * KX;
  const unsigned short* brow1 = Bt + (size_t)(tN * 32 + 16 + ln) * KX;

#pragma unroll 4
  for (int kk = 0; kk < KX / 32; ++kk) {
    V16U af;   // A layout: row = lane%16; elem i -> k = i+16*hi
    af.h[0] = *(const v8bf*)(arow + kk * 32 + hi * 16);
    af.h[1] = *(const v8bf*)(arow + kk * 32 + hi * 16 + 8);
    const v16bf b0 = *(const v16bf*)(brow0 + kk * 32 + hi * 16);
    const v16bf b1 = *(const v16bf*)(brow1 + kk * 32 + hi * 16);
    acc[0] = __builtin_amdgcn_wmma_f32_16x16x32_bf16(false, af.v, false, b0,
                                                     (short)0, acc[0], false, false);
    acc[1] = __builtin_amdgcn_wmma_f32_16x16x32_bf16(false, af.v, false, b1,
                                                     (short)0, acc[1], false, false);
  }

#pragma unroll
  for (int nt = 0; nt < 2; ++nt) {
#pragma unroll
    for (int r = 0; r < 8; ++r) {
      const int row = tM * 16 + r + hi * 8;
      const int col = tN * 32 + nt * 16 + ln;
      const float v = acc[nt][r];
      if (mode == 2) {
        ((float*)Cout)[(size_t)row * rowStride + col] = v;
      } else if (mode == 0) {
        ((unsigned short*)Cout)[(size_t)row * rowStride + col] = f2bf_bits(v);
      } else {  // V transposed: row = b*1024+s, col = g*96+d
        ((unsigned short*)Cout)[((size_t)((row >> 10) * (NG * DHEAD) + col)) * S_ +
                                (row & (S_ - 1))] = f2bf_bits(v);
      }
    }
  }
}

// =====================================================================
// Flash attention: 8 waves/block = 128 query rows of one (b, head).
// K (32x96) and V^T (96x32) tiles are staged in LDS once per S-step,
// double-buffered via async global->LDS DMA (ASYNCcnt) so tile i+1
// streams in while tile i runs 12 bf16 WMMAs. Per-wave causal skip.
// =====================================================================
__global__ void __launch_bounds__(256)
flash_attn_kernel(const unsigned short* __restrict__ Qb,
                  const unsigned short* __restrict__ Kb,
                  const unsigned short* __restrict__ Vtb,
                  unsigned short* __restrict__ AVcat) {
  __shared__ __align__(32) unsigned short Kt[2][32][DHEAD];   // 2 x 6KB
  __shared__ __align__(32) unsigned short Vt2[2][DHEAD][32];  // 2 x 6KB
  __shared__ __align__(32) unsigned short Pst[8][16][32];     // per-wave P staging

  const int tid = threadIdx.x;
  const int lane = tid & 31;
  const int w = tid >> 5;
  const int hi = lane >> 4, ln = lane & 15;
  const int H = blockIdx.y, b = blockIdx.z, g = H & (NG - 1);
  const int t0 = blockIdx.x * 128 + w * 16;

  // cooperative staging of one S-step (32 keys) into LDS buffer `buf`
  auto stage = [&](int buf, int s0) {
    for (int c = tid; c < 32 * (DHEAD / 8); c += 256) {     // 384 x 16B: K tile
      int row = c / (DHEAD / 8), part = c % (DHEAD / 8);
      async_copy16(Kb + (size_t)(b * S_ + s0 + row) * (NG * DHEAD) + g * DHEAD + part * 8,
                   &Kt[buf][row][part * 8]);
    }
    for (int c = tid; c < DHEAD * 4; c += 256) {            // 384 x 16B: V^T tile
      int d = c >> 2, part = c & 3;
      async_copy16(Vtb + ((size_t)(b * NG + g) * DHEAD + d) * S_ + s0 + part * 8,
                   &Vt2[buf][d][part * 8]);
    }
  };

  // ---- Q A-fragments: 3 k-steps covering the 96-wide contraction ----
  V16U qf[3];
  {
    const unsigned short* qrow =
        Qb + (size_t)(b * T_ + (t0 + ln)) * (NH * DHEAD) + H * DHEAD;
#pragma unroll
    for (int kk = 0; kk < 3; ++kk) {
      int k0 = kk * 32 + hi * 16;
      qf[kk].h[0] = *(const v8bf*)(qrow + k0);
      qf[kk].h[1] = *(const v8bf*)(qrow + k0 + 8);
    }
  }

  v8f Oacc[6];
#pragma unroll
  for (int n = 0; n < 6; ++n)
#pragma unroll
    for (int r = 0; r < 8; ++r) Oacc[n][r] = 0.f;
  float mrow[8], lrow[8];
#pragma unroll
  for (int r = 0; r < 8; ++r) { mrow[r] = -1.0e30f; lrow[r] = 0.f; }

  const int nBlk = blockIdx.x * 4 + 4;      // uniform causal trip count/block
  stage(0, 0);

  for (int st = 0; st < nBlk; ++st) {
    const int buf = st & 1;
    const int s0 = st * 32;

    wait_async_all();          // my wave's async chunks for tile st landed
    __syncthreads();           // everyone's chunks landed
    if (st + 1 < nBlk) stage(buf ^ 1, s0 + 32);   // overlap DMA with compute

    if (s0 < t0 + 16) {        // wave-level causal skip (EXEC stays all-ones)
      // ---- logits: 2 n-tiles x 3 k-steps of WMMA from LDS K tile ----
      v8f cfr[2];
#pragma unroll
      for (int nt = 0; nt < 2; ++nt) {
#pragma unroll
        for (int r = 0; r < 8; ++r) cfr[nt][r] = 0.f;
#pragma unroll
        for (int kk = 0; kk < 3; ++kk) {
          const v16bf bfrag =
              *(const v16bf*)(&Kt[buf][nt * 16 + ln][kk * 32 + hi * 16]);
          cfr[nt] = __builtin_amdgcn_wmma_f32_16x16x32_bf16(
              false, qf[kk].v, false, bfrag, (short)0, cfr[nt], false, false);
        }
      }
      // ---- causal mask ----
#pragma unroll
      for (int nt = 0; nt < 2; ++nt) {
        const int scol = s0 + nt * 16 + ln;
#pragma unroll
        for (int r = 0; r < 8; ++r)
          if (scol > t0 + r + hi * 8) cfr[nt][r] = -1.0e30f;
      }
      // ---- online softmax (row lives in one 16-lane half) ----
      float mnew[8], alpha[8];
#pragma unroll
      for (int r = 0; r < 8; ++r) {
        float v = fmaxf(cfr[0][r], cfr[1][r]);
        v = fmaxf(v, __shfl_xor(v, 1, 32));
        v = fmaxf(v, __shfl_xor(v, 2, 32));
        v = fmaxf(v, __shfl_xor(v, 4, 32));
        v = fmaxf(v, __shfl_xor(v, 8, 32));
        mnew[r] = fmaxf(mrow[r], v);
        alpha[r] = __expf(mrow[r] - mnew[r]);
        mrow[r] = mnew[r];
      }
#pragma unroll
      for (int nt = 0; nt < 2; ++nt)
#pragma unroll
        for (int r = 0; r < 8; ++r) cfr[nt][r] = __expf(cfr[nt][r] - mnew[r]);
#pragma unroll
      for (int r = 0; r < 8; ++r) {
        float rs = cfr[0][r] + cfr[1][r];
        rs += __shfl_xor(rs, 1, 32);
        rs += __shfl_xor(rs, 2, 32);
        rs += __shfl_xor(rs, 4, 32);
        rs += __shfl_xor(rs, 8, 32);
        lrow[r] = lrow[r] * alpha[r] + rs;
      }
#pragma unroll
      for (int n = 0; n < 6; ++n)
#pragma unroll
        for (int r = 0; r < 8; ++r) Oacc[n][r] *= alpha[r];

      // ---- transpose P (D-layout f32) -> A-layout bf16 via wave LDS ----
#pragma unroll
      for (int nt = 0; nt < 2; ++nt)
#pragma unroll
        for (int r = 0; r < 8; ++r)
          Pst[w][r + hi * 8][nt * 16 + ln] = f2bf_bits(cfr[nt][r]);
      V16U pf;
      {
        const unsigned short* prow = &Pst[w][ln][0];
        pf.h[0] = *(const v8bf*)(prow + hi * 16);
        pf.h[1] = *(const v8bf*)(prow + hi * 16 + 8);
      }
      // ---- P(16x32) * V(32x96): 6 d-tiles from LDS V^T tile ----
#pragma unroll
      for (int dt = 0; dt < 6; ++dt) {
        const v16bf vfrag = *(const v16bf*)(&Vt2[buf][dt * 16 + ln][hi * 16]);
        Oacc[dt] = __builtin_amdgcn_wmma_f32_16x16x32_bf16(
            false, pf.v, false, vfrag, (short)0, Oacc[dt], false, false);
      }
    }
    __syncthreads();           // all reads of buf done before it is refilled
  }

  // ---- finalize: /L, store bf16 AVcat[bt][ stream*512 + H*32 + hv ] ----
#pragma unroll
  for (int r = 0; r < 8; ++r) {
    const float inv = 1.0f / lrow[r];
    const int t = t0 + r + hi * 8;
    unsigned short* avrow = AVcat + (size_t)(b * T_ + t) * (3 * KX);
#pragma unroll
    for (int dt = 0; dt < 6; ++dt) {
      const int d = dt * 16 + ln;
      avrow[(d >> 5) * KX + H * 32 + (d & 31)] = f2bf_bits(Oacc[dt][r] * inv);
    }
  }
}

// =====================================================================
extern "C" void kernel_launch(void* const* d_in, const int* in_sizes, int n_in,
                              void* d_out, int out_size, void* d_ws, size_t ws_size,
                              hipStream_t stream) {
  (void)in_sizes; (void)n_in; (void)out_size; (void)ws_size;
  const float* target_inv = (const float*)d_in[0];
  const float* target_vec = (const float*)d_in[1];
  const float* source_inv = (const float*)d_in[2];
  const float* source_vec = (const float*)d_in[3];
  const float* Wq_inv = (const float*)d_in[4];
  const float* Wq_vec = (const float*)d_in[5];
  const float* Wk_inv = (const float*)d_in[6];
  const float* Wk_vec = (const float*)d_in[7];
  const float* Wv_inv = (const float*)d_in[8];
  const float* Wv_vec = (const float*)d_in[9];
  const float* Wo_inv = (const float*)d_in[10];
  const float* Wo_vec = (const float*)d_in[11];
  // d_in[12] = mask: deterministic causal tril, encoded in flash kernel.

  const int MROWS = B_ * T_;                 // 2048
  // ---- workspace layout (bf16 unless noted), ~21 MB total ----
  unsigned short* Xt     = (unsigned short*)d_ws;          // 2048x512
  unsigned short* Xs     = Xt     + (size_t)MROWS * KX;    // 2048x512
  unsigned short* WqCt   = Xs     + (size_t)MROWS * KX;    // 1536x512
  unsigned short* WkCt   = WqCt   + (size_t)NH * DHEAD * KX;   // 384x512
  unsigned short* WvCt   = WkCt   + (size_t)NG * DHEAD * KX;   // 384x512
  unsigned short* WoInvT = WvCt   + (size_t)NG * DHEAD * KX;   // 256x512
  unsigned short* WoVecT = WoInvT + (size_t)CINV * KX;     // 128x512
  unsigned short* Qb     = WoVecT + (size_t)CVEC * KX;     // 2048x1536
  unsigned short* Kb     = Qb     + (size_t)MROWS * NH * DHEAD; // 2048x384
  unsigned short* Vtb    = Kb     + (size_t)MROWS * NG * DHEAD; // 384x... (transposed)
  unsigned short* AVcat  = Vtb    + (size_t)MROWS * NG * DHEAD; // 2048x1536

  // ---- prep: bf16 packing + combined transposed weights ----
  prep_x_kernel<<<(2 * MROWS * KX) / 256, 256, 0, stream>>>(
      target_inv, target_vec, source_inv, source_vec, Xt, Xs);
  prep_wq_kernel<<<(NH * DHEAD * KX) / 256, 256, 0, stream>>>(Wq_inv, Wq_vec, WqCt);
  prep_wkv_kernel<<<(NG * DHEAD * KX) / 256, 256, 0, stream>>>(
      Wk_inv, Wk_vec, Wv_inv, Wv_vec, WkCt, WvCt);
  prep_wo_kernel<<<((CINV + CVEC) * KX) / 256, 256, 0, stream>>>(
      Wo_inv, Wo_vec, WoInvT, WoVecT);

  // ---- QKV projections on WMMA ----
  gemm_bf16_kernel<<<(MROWS / 16) * (NH * DHEAD / 32) / 8, 256, 0, stream>>>(
      Xt, KX, WqCt, NH * DHEAD / 32, Qb, 0, NH * DHEAD);
  gemm_bf16_kernel<<<(MROWS / 16) * (NG * DHEAD / 32) / 8, 256, 0, stream>>>(
      Xs, KX, WkCt, NG * DHEAD / 32, Kb, 0, NG * DHEAD);
  gemm_bf16_kernel<<<(MROWS / 16) * (NG * DHEAD / 32) / 8, 256, 0, stream>>>(
      Xs, KX, WvCt, NG * DHEAD / 32, Vtb, 1, 0);

  // ---- causal flash attention ----
  dim3 grid(T_ / 128, NH, B_);
  flash_attn_kernel<<<grid, 256, 0, stream>>>(Qb, Kb, Vtb, AVcat);

  // ---- output projections on WMMA (f32 straight into d_out) ----
  const int n_inv = B_ * T_ * CINV;
  gemm_bf16_kernel<<<(MROWS / 16) * (CINV / 32) / 8, 256, 0, stream>>>(
      AVcat, 3 * KX, WoInvT, CINV / 32, d_out, 2, CINV);
  gemm_bf16_kernel<<<(MROWS / 16) * (CVEC / 32) / 8, 256, 0, stream>>>(
      AVcat + KX, 3 * KX, WoVecT, CVEC / 32,
      (float*)d_out + n_inv, 2, FVEC * CVEC);
  gemm_bf16_kernel<<<(MROWS / 16) * (CVEC / 32) / 8, 256, 0, stream>>>(
      AVcat + 2 * KX, 3 * KX, WoVecT, CVEC / 32,
      (float*)d_out + n_inv + CVEC, 2, FVEC * CVEC);
}
